// NeuralTensorFactorization_22136261443921
// MI455X (gfx1250) — compile-verified, matrix operand order checked by reading, not run on hardware
//
#include <hip/hip_runtime.h>
#include <cstdint>

#define B_BATCH   16384
#define D_DIM     256
#define NUM_USERS 5000
#define NUM_ITEMS 50000
#define STEP_N    8

typedef _Float16 h16v __attribute__((ext_vector_type(16)));
typedef _Float16 h8v  __attribute__((ext_vector_type(8)));
typedef float    f8v  __attribute__((ext_vector_type(8)));

// ---------------------------------------------------------------------------
// Tiled WMMA GEMM:  C[16384 x N] = A1[16384 x K1] @ Bt1^T + A2 @ Bt2^T + bias
// mode 0: store f32 to Cf (ldc)    mode 1: sigmoid -> f16 store to Ch (ldc)
// Tile: 128x128 per block, BK=32, 8 waves, each wave 32x64 (2x4 WMMA frags).
// Staging: double-buffered LDS fed by GLOBAL_LOAD_ASYNC_TO_LDS_B128 (ASYNCcnt)
// so the chunk-(ci+1) DMA overlaps the chunk-ci WMMA work.
// ---------------------------------------------------------------------------
__global__ __launch_bounds__(256)
void gemm_f16_wmma(const _Float16* __restrict__ A1, const _Float16* __restrict__ Bt1, int K1,
                   const _Float16* __restrict__ A2, const _Float16* __restrict__ Bt2, int K2,
                   const float* __restrict__ bias,
                   float* __restrict__ Cf, _Float16* __restrict__ Ch,
                   int ldc, int mode)
{
    __shared__ _Float16 As[2][128 * 40];   // +8 half pad: conflict-free b128
    __shared__ _Float16 Bs[2][128 * 40];

    const int tid  = threadIdx.x;
    const int lane = tid & 31;
    const int wid  = tid >> 5;
    const int wm   = (wid & 3) * 32;
    const int wn   = (wid >> 2) * 64;
    const size_t m0 = (size_t)blockIdx.y * 128;
    const size_t n0 = (size_t)blockIdx.x * 128;

    f8v acc[2][4] = {};

    const int r    = tid >> 1;
    const int kh   = (tid & 1) * 16;
    const int arow = lane & 15;
    const int koff = (lane >> 4) * 8;

    const int nc1 = K1 >> 5;
    const int NC  = nc1 + (K2 >> 5);

    auto issue_copy = [&](int ci, int buf) {
        const _Float16* A; const _Float16* Bt; int K; int k0;
        if (ci < nc1) { A = A1; Bt = Bt1; K = K1; k0 = ci << 5; }
        else          { A = A2; Bt = Bt2; K = K2; k0 = (ci - nc1) << 5; }
        const _Float16* ga = &A [(m0 + r) * (size_t)K + k0 + kh];
        const _Float16* gb = &Bt[(n0 + r) * (size_t)K + k0 + kh];
        uint32_t la = (uint32_t)(uintptr_t)&As[buf][r * 40 + kh];
        uint32_t lb = (uint32_t)(uintptr_t)&Bs[buf][r * 40 + kh];
        asm volatile("global_load_async_to_lds_b128 %0, %1, off"
                     :: "v"(la), "v"(ga) : "memory");
        asm volatile("global_load_async_to_lds_b128 %0, %1, off offset:16"
                     :: "v"(la), "v"(ga) : "memory");
        asm volatile("global_load_async_to_lds_b128 %0, %1, off"
                     :: "v"(lb), "v"(gb) : "memory");
        asm volatile("global_load_async_to_lds_b128 %0, %1, off offset:16"
                     :: "v"(lb), "v"(gb) : "memory");
    };

    issue_copy(0, 0);
    for (int ci = 0; ci < NC; ++ci) {
        const int cur = ci & 1;
        asm volatile("s_wait_asynccnt 0x0" ::: "memory");
        __syncthreads();
        if (ci + 1 < NC) issue_copy(ci + 1, cur ^ 1);

        h16v af[2];
#pragma unroll
        for (int i = 0; i < 2; ++i) {
            const _Float16* base = &As[cur][(wm + 16 * i + arow) * 40 + koff];
            h8v lo = *(const h8v*)(base);
            h8v hi = *(const h8v*)(base + 16);
            af[i] = __builtin_shufflevector(lo, hi, 0,1,2,3,4,5,6,7,8,9,10,11,12,13,14,15);
        }
        h16v bf[4];
#pragma unroll
        for (int j = 0; j < 4; ++j) {
            const _Float16* base = &Bs[cur][(wn + 16 * j + arow) * 40 + koff];
            h8v lo = *(const h8v*)(base);
            h8v hi = *(const h8v*)(base + 16);
            bf[j] = __builtin_shufflevector(lo, hi, 0,1,2,3,4,5,6,7,8,9,10,11,12,13,14,15);
        }
#pragma unroll
        for (int i = 0; i < 2; ++i)
#pragma unroll
            for (int j = 0; j < 4; ++j)
                acc[i][j] = __builtin_amdgcn_wmma_f32_16x16x32_f16(
                    false, af[i], false, bf[j], (short)0, acc[i][j], false, false);
    }

    const int n    = lane & 15;
    const int rsel = (lane >> 4) * 8;
#pragma unroll
    for (int j = 0; j < 4; ++j) {
        const size_t col = n0 + wn + j * 16 + n;
        const float bv = bias ? bias[col] : 0.0f;
#pragma unroll
        for (int i = 0; i < 2; ++i) {
#pragma unroll
            for (int v = 0; v < 8; ++v) {
                const size_t row = m0 + wm + i * 16 + rsel + v;
                float val = acc[i][j][v] + bv;
                if (mode == 0) Cf[row * (size_t)ldc + col] = val;
                else           Ch[row * (size_t)ldc + col] =
                                   (_Float16)(1.0f / (1.0f + __expf(-val)));
            }
        }
    }
}

__global__ void zero_kernel(uint32_t* p, int nwords) {
    int id = blockIdx.x * 256 + threadIdx.x;
    if (id < nwords) p[id] = 0u;
}

__global__ void tconv_kernel(const float* __restrict__ in, _Float16* __restrict__ out,
                             int R, int C) {
    int id = blockIdx.x * 256 + threadIdx.x;
    if (id >= R * C) return;
    int rr = id / C, cc = id % C;
    out[(size_t)cc * R + rr] = (_Float16)in[id];
}

__global__ void gather_xt_kernel(const int* __restrict__ x, const float* __restrict__ emb,
                                 _Float16* __restrict__ xt, int step) {
    int id = blockIdx.x * 256 + threadIdx.x;
    int b = id >> 8, d = id & 255;
    int t = x[b * 3];
    size_t row = (size_t)(NUM_USERS + NUM_ITEMS) + t + step;
    xt[id] = (_Float16)emb[row * 256 + d];
}

__global__ void gather_ui_kernel(const int* __restrict__ x, const float* __restrict__ emb,
                                 _Float16* __restrict__ z) {
    int id = blockIdx.x * 256 + threadIdx.x;
    int b = id >> 9, j = id & 511;
    int row, d;
    if (j < 256) { row = x[b * 3 + 1];             d = j; }
    else         { row = x[b * 3 + 2] + NUM_USERS; d = j - 256; }
    z[(size_t)b * 768 + j] = (_Float16)emb[(size_t)row * 256 + d];
}

__global__ void sub_update_kernel(const float* __restrict__ agg, _Float16* __restrict__ sub,
                                  _Float16* __restrict__ scat, const float* __restrict__ mix,
                                  const float* __restrict__ coef) {
    int id = blockIdx.x * 256 + threadIdx.x;
    int b = id >> 8, d = id & 255;
    float a0 = agg[id];
    float a1 = agg[B_BATCH * D_DIM + id];

    float xx = fminf(fmaxf(a0, -1.0f), 1.0f);
    float kb[11];
#pragma unroll
    for (int i = 0; i < 11; ++i) {
        float k0 = (i - 3) * 0.4f - 1.0f;
        float k1 = (i - 2) * 0.4f - 1.0f;
        kb[i] = (xx >= k0 && xx < k1) ? 1.0f : 0.0f;
    }
#pragma unroll
    for (int dg = 1; dg <= 3; ++dg) {
        float inv = 1.0f / (dg * 0.4f);
#pragma unroll
        for (int i = 0; i < 8; ++i) {
            float ki   = (i - 3) * 0.4f - 1.0f;
            float kid1 = (i + dg - 2) * 0.4f - 1.0f;
            kb[i] = (xx - ki) * inv * kb[i] + (kid1 - xx) * inv * kb[i + 1];
        }
        for (int i = 8; i < 11 - dg; ++i) {
            float ki   = (i - 3) * 0.4f - 1.0f;
            float kid1 = (i + dg - 2) * 0.4f - 1.0f;
            kb[i] = (xx - ki) * inv * kb[i] + (kid1 - xx) * inv * kb[i + 1];
        }
    }
    float s0 = 0.0f;
#pragma unroll
    for (int i = 0; i < 8; ++i) s0 += kb[i] * coef[i];
    float s1 = a1 * a1;

    scat[(size_t)b * 512 + d]       = (_Float16)s0;
    scat[(size_t)b * 512 + 256 + d] = (_Float16)s1;

    float old0 = (float)sub[id];
    float old1 = (float)sub[B_BATCH * D_DIM + id];
    sub[id]                   = (_Float16)(mix[d]       * s0 + mix[256 + d] * old0);
    sub[B_BATCH * D_DIM + id] = (_Float16)(mix[512 + d] * s1 + mix[768 + d] * old1);
}

__global__ void cell_update_kernel(const float* __restrict__ gates, const float* __restrict__ opre,
                                   float* __restrict__ c, _Float16* __restrict__ h) {
    int id = blockIdx.x * 256 + threadIdx.x;
    int b = id >> 8, d = id & 255;
    size_t gb = (size_t)b * 768;
    float ig = 1.0f / (1.0f + __expf(-gates[gb + d]));
    float fg = 1.0f / (1.0f + __expf(-gates[gb + 256 + d]));
    float g  = tanhf(gates[gb + 512 + d]);
    float o  = 1.0f / (1.0f + __expf(-opre[id]));
    float cn = fg * c[id] + ig * g;
    c[id] = cn;
    h[id] = (_Float16)(o * tanhf(cn));
}

__global__ __launch_bounds__(256)
void bn_relu_kernel(const float* __restrict__ Z, const float* __restrict__ gamma,
                    const float* __restrict__ beta, _Float16* __restrict__ outH,
                    float* __restrict__ outF, int ncols) {
    __shared__ float red[512];
    const int col = blockIdx.x;
    const int tid = threadIdx.x;
    float s = 0.0f, s2 = 0.0f;
    for (int r0 = tid; r0 < B_BATCH; r0 += 256) {
        float v = Z[(size_t)r0 * ncols + col];
        s += v; s2 += v * v;
    }
    red[tid] = s; red[256 + tid] = s2;
    __syncthreads();
    for (int st = 128; st > 0; st >>= 1) {
        if (tid < st) { red[tid] += red[tid + st]; red[256 + tid] += red[256 + tid + st]; }
        __syncthreads();
    }
    const float mu  = red[0]   * (1.0f / B_BATCH);
    const float var = red[256] * (1.0f / B_BATCH) - mu * mu;
    const float inv = rsqrtf(var + 1e-5f);
    const float ga = gamma[col], be = beta[col];
    for (int r0 = tid; r0 < B_BATCH; r0 += 256) {
        float v = (Z[(size_t)r0 * ncols + col] - mu) * inv * ga + be;
        v = fmaxf(v, 0.0f);
        if (outH) outH[(size_t)r0 * ncols + col] = (_Float16)v;
        else      outF[(size_t)r0 * ncols + col] = v;
    }
}

__global__ __launch_bounds__(256)
void fc_kernel(const float* __restrict__ Z, const float* __restrict__ w,
               const float* __restrict__ bptr, float* __restrict__ out) {
    int wid = threadIdx.x >> 5, lane = threadIdx.x & 31;
    int row = blockIdx.x * 8 + wid;
    const float* zr = Z + (size_t)row * 256;
    float s = 0.0f;
#pragma unroll
    for (int k = lane; k < 256; k += 32) s += zr[k] * w[k];
#pragma unroll
    for (int off = 16; off > 0; off >>= 1) s += __shfl_xor(s, off, 32);
    if (lane == 0) out[row] = s + bptr[0];
}

extern "C" void kernel_launch(void* const* d_in, const int* in_sizes, int n_in,
                              void* d_out, int out_size, void* d_ws, size_t ws_size,
                              hipStream_t stream)
{
    (void)in_sizes; (void)n_in; (void)out_size; (void)ws_size;
    const int*   x      = (const int*)  d_in[0];
    const float* emb    = (const float*)d_in[1];
    const float* Wg     = (const float*)d_in[2];
    const float* Ug     = (const float*)d_in[3];
    const float* bg     = (const float*)d_in[4];
    const float* subWx  = (const float*)d_in[5];
    const float* subWh  = (const float*)d_in[6];
    const float* submix = (const float*)d_in[7];
    const float* scoef  = (const float*)d_in[8];
    const float* aggW   = (const float*)d_in[9];
    const float* aggb   = (const float*)d_in[10];
    const float* projW  = (const float*)d_in[11];
    const float* projb  = (const float*)d_in[12];
    const float* W0     = (const float*)d_in[13];
    const float* b0     = (const float*)d_in[14];
    const float* gamma0 = (const float*)d_in[15];
    const float* beta0  = (const float*)d_in[16];
    const float* W1     = (const float*)d_in[17];
    const float* b1     = (const float*)d_in[18];
    const float* gamma1 = (const float*)d_in[19];
    const float* beta1  = (const float*)d_in[20];
    const float* fcW    = (const float*)d_in[21];
    const float* fcb    = (const float*)d_in[22];
    float* out = (float*)d_out;

    char* ws = (char*)d_ws;
    size_t off = 0;
    auto alloc = [&](size_t bytes) -> size_t {
        size_t r = off; off = (off + bytes + 255) & ~(size_t)255; return r;
    };
    const size_t BD = (size_t)B_BATCH * D_DIM;

    size_t oWgT   = alloc(768 * 256 * 2);
    size_t oUgT   = alloc(768 * 256 * 2);
    size_t oWxT0  = alloc(256 * 256 * 2);
    size_t oWxT1  = alloc(256 * 256 * 2);
    size_t oWhT0  = alloc(256 * 256 * 2);
    size_t oWhT1  = alloc(256 * 256 * 2);
    size_t oAggWT = alloc(256 * 512 * 2);
    size_t oProjT = alloc(256 * 256 * 2);
    size_t oW0T   = alloc(1024 * 768 * 2);
    size_t oW1T   = alloc(256 * 1024 * 2);
    size_t oH   = alloc(BD * 2);
    size_t oC   = alloc(BD * 4);
    size_t oSub = alloc(2 * BD * 2);
    size_t sbase = off, so = off;
    auto salloc = [&](size_t bytes) -> size_t {
        size_t r = so; so = (so + bytes + 255) & ~(size_t)255; return r;
    };
    size_t oXt    = salloc(BD * 2);
    size_t oGates = salloc((size_t)B_BATCH * 768 * 4);
    size_t oAgg   = salloc(2 * BD * 4);
    size_t oScat  = salloc((size_t)B_BATCH * 512 * 2);
    size_t oOpre  = salloc(BD * 4);
    so = sbase;
    size_t oZ   = salloc((size_t)B_BATCH * 768 * 2);
    size_t oZ0  = salloc((size_t)B_BATCH * 1024 * 4);
    size_t oZ0h = salloc((size_t)B_BATCH * 1024 * 2);
    size_t oZ1  = salloc(BD * 4);
    size_t oZ1n = salloc(BD * 4);

    _Float16* WgT   = (_Float16*)(ws + oWgT);
    _Float16* UgT   = (_Float16*)(ws + oUgT);
    _Float16* WxT0  = (_Float16*)(ws + oWxT0);
    _Float16* WxT1  = (_Float16*)(ws + oWxT1);
    _Float16* WhT0  = (_Float16*)(ws + oWhT0);
    _Float16* WhT1  = (_Float16*)(ws + oWhT1);
    _Float16* AggWT = (_Float16*)(ws + oAggWT);
    _Float16* ProjT = (_Float16*)(ws + oProjT);
    _Float16* W0T   = (_Float16*)(ws + oW0T);
    _Float16* W1T   = (_Float16*)(ws + oW1T);
    _Float16* Hh    = (_Float16*)(ws + oH);
    float*    Cc    = (float*)   (ws + oC);
    _Float16* Sub   = (_Float16*)(ws + oSub);
    _Float16* Xt    = (_Float16*)(ws + oXt);
    float*    Gates = (float*)   (ws + oGates);
    float*    Agg   = (float*)   (ws + oAgg);
    _Float16* Scat  = (_Float16*)(ws + oScat);
    float*    Opre  = (float*)   (ws + oOpre);
    _Float16* Zb    = (_Float16*)(ws + oZ);
    float*    Z0    = (float*)   (ws + oZ0);
    _Float16* Z0h   = (_Float16*)(ws + oZ0h);
    float*    Z1    = (float*)   (ws + oZ1);
    float*    Z1n   = (float*)   (ws + oZ1n);

    auto tconv = [&](const float* in, _Float16* outp, int R, int C) {
        int n = R * C;
        tconv_kernel<<<(n + 255) / 256, 256, 0, stream>>>(in, outp, R, C);
    };
    auto gemm = [&](const _Float16* A1, const _Float16* Bt1, int K1,
                    const _Float16* A2, const _Float16* Bt2, int K2,
                    const float* bias, float* Cf, _Float16* Ch, int ldc, int N, int mode) {
        dim3 grid(N / 128, B_BATCH / 128);
        gemm_f16_wmma<<<grid, 256, 0, stream>>>(A1, Bt1, K1, A2, Bt2, K2, bias, Cf, Ch, ldc, mode);
    };

    tconv(Wg, WgT, 256, 768);
    tconv(Ug, UgT, 256, 768);
    tconv(subWx,          WxT0, 256, 256);
    tconv(subWx + 65536,  WxT1, 256, 256);
    tconv(subWh,          WhT0, 256, 256);
    tconv(subWh + 65536,  WhT1, 256, 256);
    tconv(aggW,  AggWT, 512, 256);
    tconv(projW, ProjT, 256, 256);
    tconv(W0, W0T, 768, 1024);
    tconv(W1, W1T, 1024, 256);

    {
        int nwords = (int)((BD * 2 + BD * 4 + 2 * BD * 2) / 4);
        zero_kernel<<<(nwords + 255) / 256, 256, 0, stream>>>((uint32_t*)(ws + oH), nwords);
    }

    const int ewGrid = (int)(BD / 256);
    for (int step = 0; step < STEP_N; ++step) {
        gather_xt_kernel<<<ewGrid, 256, 0, stream>>>(x, emb, Xt, step);
        gemm(Xt, WgT, 256, Hh, UgT, 256, bg, Gates, nullptr, 768, 768, 0);
        gemm(Xt, WxT0, 256, Sub,      WhT0, 256, nullptr, Agg,      nullptr, 256, 256, 0);
        gemm(Xt, WxT1, 256, Sub + BD, WhT1, 256, nullptr, Agg + BD, nullptr, 256, 256, 0);
        sub_update_kernel<<<ewGrid, 256, 0, stream>>>(Agg, Sub, Scat, submix, scoef);
        gemm(Scat, AggWT, 512, nullptr, nullptr, 0, aggb, Opre, nullptr, 256, 256, 0);
        cell_update_kernel<<<ewGrid, 256, 0, stream>>>(Gates, Opre, Cc, Hh);
    }

    gather_ui_kernel<<<(int)(B_BATCH * 512 / 256), 256, 0, stream>>>(x, emb, Zb);
    gemm(Hh, ProjT, 256, nullptr, nullptr, 0, projb, nullptr, Zb + 512, 768, 256, 1);

    gemm(Zb, W0T, 768, nullptr, nullptr, 0, b0, Z0, nullptr, 1024, 1024, 0);
    bn_relu_kernel<<<1024, 256, 0, stream>>>(Z0, gamma0, beta0, Z0h, nullptr, 1024);
    gemm(Z0h, W1T, 1024, nullptr, nullptr, 0, b1, Z1, nullptr, 256, 256, 0);
    bn_relu_kernel<<<256, 256, 0, stream>>>(Z1, gamma1, beta1, nullptr, Z1n, 256);
    fc_kernel<<<B_BATCH / 8, 256, 0, stream>>>(Z1n, fcW, fcb, out);
}
